// MultiheadLatentAttention_81432579932271
// MI455X (gfx1250) — compile-verified, matrix operand order checked by reading, not run on hardware
//
#include <hip/hip_runtime.h>
#include <hip/hip_bf16.h>

typedef __attribute__((ext_vector_type(16))) _Float16 v16h;
typedef __attribute__((ext_vector_type(8)))  _Float16 v8h;
typedef __attribute__((ext_vector_type(8)))  float    v8f;

#define BATCH   2
#define SEQ     2048
#define HIDDIM  2048
#define NHEAD   16
#define DHEAD   128
#define VDIM    64
#define RANK    64
#define ROPE_D  64
#define BSTOK   (BATCH*SEQ)
#define SM_SCALE 0.08838834764831845f   /* 128^-0.5 */
#define LOG2E    1.4426950408889634f
#define LN_1E4   9.210340371976184f
#define TWO_PI   6.283185307179586f

// ---------------------------------------------------------------------------
// WMMA fragment helpers (CDNA5 wave32 layouts, CDNA5 ISA §7.12.2)
// A (16x32 f16): lane m=l&15 holds row m; elements 0..7 -> K=kb..kb+7,
//                elements 8..15 -> K=kb+16..kb+23 with kb=(l>>4)*8.
// B (32x16 f16): lane n=l&15 holds column n; same K permutation.
// C/D (16x16 f32): 8 VGPRs; m = r + 8*(l>=16), n = l&15.
// ---------------------------------------------------------------------------
__device__ inline v16h frag_ld_h(const _Float16* base, int row0, int ld, int k0, int lane) {
    int r  = row0 + (lane & 15);
    int kb = ((lane >> 4) & 1) * 8;
    const _Float16* p = base + (size_t)r * ld + k0 + kb;
    v8h lo = *(const v8h*)p;
    v8h hc = *(const v8h*)(p + 16);
    v16h a;
#pragma unroll
    for (int i = 0; i < 8; ++i) { a[i] = lo[i]; a[8 + i] = hc[i]; }
    return a;
}

__device__ inline v16h frag_ld_f32cvt(const float* base, int row0, int ld, int k0, int lane) {
    int r  = row0 + (lane & 15);
    int kb = ((lane >> 4) & 1) * 8;
    const float* p = base + (size_t)r * ld + k0 + kb;
    v16h a;
#pragma unroll
    for (int i = 0; i < 8; ++i) { a[i] = (_Float16)p[i]; a[8 + i] = (_Float16)p[16 + i]; }
    return a;
}

__device__ inline v8f wmma16(v16h a, v16h b, v8f c) {
    return __builtin_amdgcn_wmma_f32_16x16x32_f16(false, a, false, b, (short)0, c, false, false);
}

// Async global->LDS copy of 16 bytes per lane (CDNA5, CDNA5 ISA):
// bypasses VGPRs, tracked by ASYNCcnt (in-order completion per wave).
__device__ inline void async_ld16(unsigned lds_off, const void* gaddr) {
    asm volatile("global_load_async_to_lds_b128 %0, %1, off"
                 :: "v"(lds_off), "v"((unsigned long long)(uintptr_t)gaddr)
                 : "memory");
}

// YaRN cos/sin for rope dim 64 (i in 0..31), computed on device.
__device__ inline void yarn_cs(int pos, int i, float* cs, float* sn) {
    float expo = (float)i * (1.0f / 32.0f);
    float fe   = __expf(-expo * LN_1E4);     // 10000^-expo
    float fi   = fe * (1.0f / 40.0f);        // extrapolated / FACTOR
    float low  = floorf(64.0f * __logf(4096.0f / (32.0f * TWO_PI)) / (2.0f * LN_1E4));
    float high = ceilf (64.0f * __logf(4096.0f / ( 1.0f * TWO_PI)) / (2.0f * LN_1E4));
    low = fmaxf(low, 0.0f); high = fminf(high, 63.0f);
    float ramp = fminf(fmaxf(((float)i - low) / (high - low), 0.0f), 1.0f);
    float mask = 1.0f - ramp;
    float invf = fi * (1.0f - mask) + fe * mask;
    float msc  = 0.1f * 0.707f * __logf(40.0f) + 1.0f;
    float ang  = (float)pos * invf;
    *cs = __cosf(ang) * msc;
    *sn = __sinf(ang) * msc;
}

// ---------------------------------------------------------------------------
// Kernel 0: transpose + f32->f16 convert weights (one-time, bandwidth trivial)
// dst[C][R] = src[R][C]
// ---------------------------------------------------------------------------
__global__ void transpose_f32_to_f16_kernel(const float* __restrict__ src,
                                            _Float16* __restrict__ dst, int R, int C) {
    int idx = blockIdx.x * 256 + threadIdx.x;
    if (idx >= R * C) return;
    int c = idx / R, r = idx - c * R;
    dst[(size_t)c * R + r] = (_Float16)src[(size_t)r * C + c];
}

// ---------------------------------------------------------------------------
// Kernel 1: qa = rmsnorm(hs@wq_a), ckv = hs@wkv_a -> cnorm + roped kpe
// 1 block = 16 tokens, 4 waves, 12 output 16x16 tiles (3/wave), K=2048.
// ---------------------------------------------------------------------------
__global__ __launch_bounds__(128) void proj_a_kernel(
    const float* __restrict__ hs, const _Float16* __restrict__ wqaT,
    const _Float16* __restrict__ wkvaT, const float* __restrict__ qanw,
    const float* __restrict__ kvnw, _Float16* __restrict__ qa_n,
    _Float16* __restrict__ c_n, _Float16* __restrict__ kpe_r) {
    __shared__ __align__(16) float accf[16][192];  // [tok][qa 0:64 | ckv 64:192]
    __shared__ float rs[16][2];
    int tok0 = blockIdx.x * 16;
    int wave = threadIdx.x >> 5, lane = threadIdx.x & 31;

    v8f zero = {};
    v8f acc[3];
#pragma unroll
    for (int t = 0; t < 3; ++t) acc[t] = zero;

    for (int ks = 0; ks < HIDDIM; ks += 32) {
        v16h af = frag_ld_f32cvt(hs, tok0, HIDDIM, ks, lane);
#pragma unroll
        for (int t = 0; t < 3; ++t) {
            int tile = wave * 3 + t;
            const _Float16* WT = (tile < 4) ? wqaT : wkvaT;
            int n0 = (tile < 4) ? tile * 16 : (tile - 4) * 16;
            v16h bf = frag_ld_h(WT, n0, HIDDIM, ks, lane);
            acc[t] = wmma16(af, bf, acc[t]);
        }
    }
    {
        int n = lane & 15, hi = lane >> 4;
#pragma unroll
        for (int t = 0; t < 3; ++t) {
            int col0 = (wave * 3 + t) * 16;
#pragma unroll
            for (int r = 0; r < 8; ++r) accf[r + 8 * hi][col0 + n] = acc[t][r];
        }
    }
    __syncthreads();
    if (threadIdx.x < 16) {
        int m = threadIdx.x;
        float sq = 0.f, sk = 0.f;
        for (int c = 0; c < 64; ++c) {
            float x = accf[m][c];      sq += x * x;
            float y = accf[m][64 + c]; sk += y * y;
        }
        rs[m][0] = rsqrtf(sq * (1.0f / 64.0f) + 1e-6f);
        rs[m][1] = rsqrtf(sk * (1.0f / 64.0f) + 1e-6f);
    }
    __syncthreads();
    int s0 = tok0 & (SEQ - 1);
    for (int idx = threadIdx.x; idx < 1024; idx += 128) {
        int m = idx >> 6, c = idx & 63;
        size_t g = (size_t)(tok0 + m) * 64 + c;
        qa_n[g] = (_Float16)(accf[m][c]      * rs[m][0] * qanw[c]);
        c_n[g]  = (_Float16)(accf[m][64 + c] * rs[m][1] * kvnw[c]);
        // rope on k_pe (interleave-permuted then rotate-half, per reference)
        float out, cs, sn;
        int pos = s0 + m;
        if (c < 32) {
            yarn_cs(pos, c, &cs, &sn);
            out = accf[m][128 + 2 * c] * cs - accf[m][128 + 2 * c + 1] * sn;
        } else {
            int i = c - 32;
            yarn_cs(pos, i, &cs, &sn);
            out = accf[m][128 + 2 * i + 1] * cs + accf[m][128 + 2 * i] * sn;
        }
        kpe_r[g] = (_Float16)out;
    }
}

// ---------------------------------------------------------------------------
// Kernel 2: per (token-tile, head): Q (rope on pe half), K (nope | kpe), V^T.
// 8 waves, 16 col tiles (2/wave), K=64 (2 WMMA steps).
// ---------------------------------------------------------------------------
__global__ __launch_bounds__(256) void proj_b_kernel(
    const _Float16* __restrict__ qa_n, const _Float16* __restrict__ c_n,
    const _Float16* __restrict__ kpe_r, const _Float16* __restrict__ wqbT,
    const _Float16* __restrict__ wkvbT, _Float16* __restrict__ Qb,
    _Float16* __restrict__ Kb, _Float16* __restrict__ VTb) {
    __shared__ float qpe[16][64];
    int tok0 = blockIdx.x * 16;
    int h = blockIdx.y;
    int b = tok0 >> 11;            // SEQ = 2048 tokens per batch
    int s0 = tok0 & (SEQ - 1);
    int wave = threadIdx.x >> 5, lane = threadIdx.x & 31;
    size_t bh = (size_t)b * NHEAD + h;
    int n = lane & 15, hi = lane >> 4;

    const _Float16* Asrc = (wave < 4) ? qa_n : c_n;   // waves 0-3: q, 4-7: kv
    v16h a0 = frag_ld_h(Asrc, tok0, RANK, 0, lane);
    v16h a1 = frag_ld_h(Asrc, tok0, RANK, 32, lane);

#pragma unroll
    for (int t = 0; t < 2; ++t) {
        int tile = wave * 2 + t;
        const _Float16* WT = (tile < 8) ? wqbT : wkvbT;
        int n0 = h * DHEAD + ((tile < 8) ? tile : (tile - 8)) * 16;
        v16h b0 = frag_ld_h(WT, n0, RANK, 0, lane);
        v16h b1 = frag_ld_h(WT, n0, RANK, 32, lane);
        v8f c = {};
        c = wmma16(a0, b0, c);
        c = wmma16(a1, b1, c);
        if (tile < 4) {                       // q_nope -> Q[:, 0:64]
#pragma unroll
            for (int r = 0; r < 8; ++r) {
                int m = r + 8 * hi;
                Qb[(bh * SEQ + s0 + m) * DHEAD + tile * 16 + n] = (_Float16)c[r];
            }
        } else if (tile < 8) {                // q_pe -> LDS for rope
#pragma unroll
            for (int r = 0; r < 8; ++r) qpe[r + 8 * hi][(tile - 4) * 16 + n] = c[r];
        } else if (tile < 12) {               // k_nope -> K[:, 0:64]
#pragma unroll
            for (int r = 0; r < 8; ++r) {
                int m = r + 8 * hi;
                Kb[(bh * SEQ + s0 + m) * DHEAD + (tile - 8) * 16 + n] = (_Float16)c[r];
            }
        } else {                              // v -> V^T [vdim][S]
            int vc = (tile - 12) * 16 + n;
#pragma unroll
            for (int r = 0; r < 8; ++r) {
                int m = r + 8 * hi;
                VTb[(bh * VDIM + vc) * SEQ + s0 + m] = (_Float16)c[r];
            }
        }
    }
    __syncthreads();
    for (int idx = threadIdx.x; idx < 1024; idx += 256) {
        int m = idx >> 6, c = idx & 63;
        int pos = s0 + m;
        float out, cs, sn;
        if (c < 32) {
            yarn_cs(pos, c, &cs, &sn);
            out = qpe[m][2 * c] * cs - qpe[m][2 * c + 1] * sn;
        } else {
            int i = c - 32;
            yarn_cs(pos, i, &cs, &sn);
            out = qpe[m][2 * i + 1] * cs + qpe[m][2 * i] * sn;
        }
        size_t rowg = (bh * SEQ + s0 + m) * DHEAD;
        Qb[rowg + 64 + c] = (_Float16)out;
        Kb[rowg + 64 + c] = kpe_r[(size_t)(tok0 + m) * 64 + c];  // shared roped k_pe
    }
}

// ---------------------------------------------------------------------------
// Kernel 3: flash attention. 1 block = (b, h, 64 q rows); 4 waves x 16 rows.
// K/V tiles DMA'd to LDS with global_load_async_to_lds_b128 (ASYNCcnt),
// double-buffered so tile i+1 streams while tile i runs 12 WMMAs.
// QK^T: 8 wmma/tile, online softmax (f32, exp2), P via padded LDS
// transpose -> A-frag, PV: 4 wmma/tile.
// ---------------------------------------------------------------------------
#define NKT (SEQ / 32)
__global__ __launch_bounds__(128) void attn_kernel(
    const _Float16* __restrict__ Qb, const _Float16* __restrict__ Kb,
    const _Float16* __restrict__ VTb, _Float16* __restrict__ AO) {
    __shared__ __align__(16) _Float16 ktile[2][32][DHEAD];  // 16 KB
    __shared__ __align__(16) _Float16 vtile[2][VDIM][32];   //  8 KB (V^T slices)
    __shared__ __align__(16) _Float16 ptile[4][16][40];     // per-wave, padded ld
    int h = blockIdx.y, b = blockIdx.z;
    size_t bh = (size_t)b * NHEAD + h;
    const _Float16* Qh = Qb  + bh * SEQ * DHEAD;
    const _Float16* Kh = Kb  + bh * SEQ * DHEAD;
    const _Float16* Vh = VTb + bh * VDIM * SEQ;
    int wave = threadIdx.x >> 5, lane = threadIdx.x & 31;
    int q0 = blockIdx.x * 64 + wave * 16;
    int n = lane & 15, hi = lane >> 4;

    int t = threadIdx.x;
    int krow = t >> 2, kseg = t & 3;   // K tile: 32 rows x 4 segs of 32 halves
    int vrow = t >> 1, vseg = t & 1;   // V tile: 64 rows x 2 segs of 16 halves

    // issue this thread's 6 async 16B transfers for one 32-key tile
    auto issue_tile = [&](int kt, int bufsel) {
        const _Float16* src = Kh + (size_t)(kt + krow) * DHEAD + kseg * 32;
        unsigned kofs = (unsigned)(uintptr_t)&ktile[bufsel][krow][kseg * 32];
#pragma unroll
        for (int j = 0; j < 4; ++j) async_ld16(kofs + j * 16, src + j * 8);
        const _Float16* vsrc = Vh + (size_t)vrow * SEQ + kt + vseg * 16;
        unsigned vofs = (unsigned)(uintptr_t)&vtile[bufsel][vrow][vseg * 16];
#pragma unroll
        for (int j = 0; j < 2; ++j) async_ld16(vofs + j * 16, vsrc + j * 8);
    };

    v16h qf[4];
#pragma unroll
    for (int kk = 0; kk < 4; ++kk) qf[kk] = frag_ld_h(Qh, q0, DHEAD, kk * 32, lane);

    v8f o0 = {}, o1 = {}, o2 = {}, o3 = {};
    float mrow[8], lrow[8];
#pragma unroll
    for (int r = 0; r < 8; ++r) { mrow[r] = -1e30f; lrow[r] = 0.f; }
    const float sc = SM_SCALE * LOG2E;

    issue_tile(0, 0);
    issue_tile(32, 1);

    for (int it = 0; it < NKT; ++it) {
        // tiles it (6 ops) and it+1 (6 ops) are outstanding, in-order:
        // asynccnt<=6 means tile `it` has fully landed.
        if (it + 1 < NKT) asm volatile("s_wait_asynccnt 0x6" ::: "memory");
        else              asm volatile("s_wait_asynccnt 0x0" ::: "memory");
        __syncthreads();
        int bs = it & 1;

        v8f s0 = {}, s1 = {};
#pragma unroll
        for (int kk = 0; kk < 4; ++kk) {
            v16h kb0 = frag_ld_h(&ktile[bs][0][0], 0,  DHEAD, kk * 32, lane);
            v16h kb1 = frag_ld_h(&ktile[bs][0][0], 16, DHEAD, kk * 32, lane);
            s0 = wmma16(qf[kk], kb0, s0);
            s1 = wmma16(qf[kk], kb1, s1);
        }
#pragma unroll
        for (int r = 0; r < 8; ++r) {
            float v0 = s0[r] * sc, v1 = s1[r] * sc;
            float rm = fmaxf(v0, v1);
#pragma unroll
            for (int off = 1; off < 16; off <<= 1) rm = fmaxf(rm, __shfl_xor(rm, off, 16));
            float mn = fmaxf(mrow[r], rm);
            float alpha = exp2f(mrow[r] - mn);
            float p0 = exp2f(v0 - mn), p1 = exp2f(v1 - mn);
            float rsum = p0 + p1;
#pragma unroll
            for (int off = 1; off < 16; off <<= 1) rsum += __shfl_xor(rsum, off, 16);
            mrow[r] = mn;
            lrow[r] = lrow[r] * alpha + rsum;
            o0[r] *= alpha; o1[r] *= alpha; o2[r] *= alpha; o3[r] *= alpha;
            ptile[wave][r + 8 * hi][n]      = (_Float16)p0;
            ptile[wave][r + 8 * hi][16 + n] = (_Float16)p1;
        }
        // P(16x32) x V(32x64)
        v16h pf  = frag_ld_h(&ptile[wave][0][0], 0, 40, 0, lane);
        v16h vb0 = frag_ld_h(&vtile[bs][0][0], 0,  32, 0, lane);
        v16h vb1 = frag_ld_h(&vtile[bs][0][0], 16, 32, 0, lane);
        v16h vb2 = frag_ld_h(&vtile[bs][0][0], 32, 32, 0, lane);
        v16h vb3 = frag_ld_h(&vtile[bs][0][0], 48, 32, 0, lane);
        o0 = wmma16(pf, vb0, o0);
        o1 = wmma16(pf, vb1, o1);
        o2 = wmma16(pf, vb2, o2);
        o3 = wmma16(pf, vb3, o3);
        __syncthreads();   // all waves done with buffer bs before refilling it
        if (it + 2 < NKT) issue_tile((it + 2) * 32, bs);
    }
#pragma unroll
    for (int r = 0; r < 8; ++r) {
        int m = r + 8 * hi;
        float inv = 1.0f / lrow[r];
        size_t rowg = (size_t)(b * SEQ + q0 + m) * (NHEAD * VDIM) + (size_t)h * VDIM;
        AO[rowg +  0 + n] = (_Float16)(o0[r] * inv);
        AO[rowg + 16 + n] = (_Float16)(o1[r] * inv);
        AO[rowg + 32 + n] = (_Float16)(o2[r] * inv);
        AO[rowg + 48 + n] = (_Float16)(o3[r] * inv);
    }
}

// ---------------------------------------------------------------------------
// Kernel 4: out = AO(f16) @ wo -> f32. 8 waves, 1 tile/wave, K=1024.
// ---------------------------------------------------------------------------
__global__ __launch_bounds__(256) void outproj_kernel(
    const _Float16* __restrict__ AO, const _Float16* __restrict__ woT,
    float* __restrict__ out) {
    int wave = threadIdx.x >> 5, lane = threadIdx.x & 31;
    int m0 = blockIdx.x * 16;
    int n0 = blockIdx.y * 128 + wave * 16;
    v8f c = {};
    for (int ks = 0; ks < NHEAD * VDIM; ks += 32) {
        v16h a  = frag_ld_h(AO,  m0, NHEAD * VDIM, ks, lane);
        v16h bf = frag_ld_h(woT, n0, NHEAD * VDIM, ks, lane);
        c = wmma16(a, bf, c);
    }
    int n = lane & 15, hi = lane >> 4;
#pragma unroll
    for (int r = 0; r < 8; ++r) {
        int m = r + 8 * hi;
        out[(size_t)(m0 + m) * HIDDIM + n0 + n] = c[r];
    }
}

// ---------------------------------------------------------------------------
extern "C" void kernel_launch(void* const* d_in, const int* in_sizes, int n_in,
                              void* d_out, int out_size, void* d_ws, size_t ws_size,
                              hipStream_t stream) {
    const float* hs   = (const float*)d_in[0];
    const float* wqa  = (const float*)d_in[1];
    const float* qanw = (const float*)d_in[2];
    const float* wqb  = (const float*)d_in[3];
    const float* wkva = (const float*)d_in[4];
    const float* kvnw = (const float*)d_in[5];
    const float* wkvb = (const float*)d_in[6];
    const float* wo   = (const float*)d_in[7];
    float* out = (float*)d_out;

    _Float16* ws = (_Float16*)d_ws;
    size_t off = 0;
    _Float16* wqaT  = ws + off; off += (size_t)RANK * HIDDIM;              // [64][2048]
    _Float16* wkvaT = ws + off; off += (size_t)(RANK + ROPE_D) * HIDDIM;   // [128][2048]
    _Float16* wqbT  = ws + off; off += (size_t)(NHEAD * DHEAD) * RANK;     // [2048][64]
    _Float16* wkvbT = ws + off; off += (size_t)(NHEAD * DHEAD) * RANK;     // [2048][64]
    _Float16* woT   = ws + off; off += (size_t)HIDDIM * (NHEAD * VDIM);    // [2048][1024]
    _Float16* qa_n  = ws + off; off += (size_t)BSTOK * RANK;
    _Float16* c_n   = ws + off; off += (size_t)BSTOK * RANK;
    _Float16* kpe_r = ws + off; off += (size_t)BSTOK * ROPE_D;
    _Float16* Qb    = ws + off; off += (size_t)BATCH * NHEAD * SEQ * DHEAD;
    _Float16* Kb    = ws + off; off += (size_t)BATCH * NHEAD * SEQ * DHEAD;
    _Float16* VTb   = ws + off; off += (size_t)BATCH * NHEAD * VDIM * SEQ;
    _Float16* AO    = ws + off; off += (size_t)BSTOK * NHEAD * VDIM;

    auto tlaunch = [&](const float* s, _Float16* d, int R, int C) {
        int nel = R * C;
        transpose_f32_to_f16_kernel<<<(nel + 255) / 256, 256, 0, stream>>>(s, d, R, C);
    };
    tlaunch(wqa,  wqaT,  HIDDIM, RANK);            // -> [64][2048]
    tlaunch(wkva, wkvaT, HIDDIM, RANK + ROPE_D);   // -> [128][2048]
    tlaunch(wqb,  wqbT,  RANK, NHEAD * DHEAD);     // -> [2048][64]
    tlaunch(wkvb, wkvbT, RANK, NHEAD * DHEAD);     // -> [2048][64]
    tlaunch(wo,   woT,   NHEAD * VDIM, HIDDIM);    // -> [2048][1024]

    proj_a_kernel<<<BSTOK / 16, 128, 0, stream>>>(hs, wqaT, wkvaT, qanw, kvnw,
                                                  qa_n, c_n, kpe_r);
    dim3 gb(BSTOK / 16, NHEAD);
    proj_b_kernel<<<gb, 256, 0, stream>>>(qa_n, c_n, kpe_r, wqbT, wkvbT, Qb, Kb, VTb);
    dim3 ga(SEQ / 64, NHEAD, BATCH);
    attn_kernel<<<ga, 128, 0, stream>>>(Qb, Kb, VTb, AO);
    dim3 go(BSTOK / 16, HIDDIM / 128);
    outproj_kernel<<<go, 256, 0, stream>>>(AO, woT, out);
}
